// GeneGAT_89060441850010
// MI455X (gfx1250) — matrix operand order repaired and checked by reading, not compiled, not run on hardware
//
#include <hip/hip_runtime.h>

#define NN 100000
#define EE 800000
#define BB 64
#define ETOT (EE + NN)
#define MT (NN / 16)   // 6250 row tiles, exact

typedef __attribute__((ext_vector_type(16))) __bf16 v16bf;
typedef __attribute__((ext_vector_type(8)))  float  v8f;

__device__ __forceinline__ unsigned f2ord(float f) {
    unsigned u = __float_as_uint(f);
    return (u & 0x80000000u) ? ~u : (u | 0x80000000u);
}
__device__ __forceinline__ float ord2f(unsigned u) {
    return (u & 0x80000000u) ? __uint_as_float(u & 0x7fffffffu) : __uint_as_float(~u);
}

// ---------------- generic fill ----------------
__global__ void k_fill(float* p, float v, int n) {
    int i = blockIdx.x * blockDim.x + threadIdx.x;
    if (i < n) p[i] = v;
}

// ---------------- input layer: h = relu(LN(x*W_in + b_in)) ----------------
__global__ __launch_bounds__(128) void k_input(const float* __restrict__ x,
        const float* __restrict__ Wi, const float* __restrict__ bi,
        const float* __restrict__ g, const float* __restrict__ b,
        float* __restrict__ h) {
    int n = blockIdx.x, j = threadIdx.x;
    __shared__ float red[128];
    float v = x[n] * Wi[j] + bi[j];
    red[j] = v; __syncthreads();
    for (int s = 64; s > 0; s >>= 1) { if (j < s) red[j] += red[j + s]; __syncthreads(); }
    float m = red[0] * (1.f / 128.f);
    __syncthreads();
    float d = v - m;
    red[j] = d * d; __syncthreads();
    for (int s = 64; s > 0; s >>= 1) { if (j < s) red[j] += red[j + s]; __syncthreads(); }
    float var = red[0] * (1.f / 128.f);
    float o = d * rsqrtf(var + 1e-5f) * g[j] + b[j];
    h[(size_t)n * 128 + j] = o > 0.f ? o : 0.f;
}

// ---------------- bf16 WMMA GEMM: C[M,Nc] = A[M,K=128] @ W[128,Nc] (+bias) ----------------
// Each wave computes a 16x64 tile: A fragment loaded once per K-chunk, 4 independent
// WMMAs per chunk, fully unrolled over K (16 v_wmma per kernel).
template <int KDIM>
__global__ __launch_bounds__(256) void k_gemm(const float* __restrict__ A,
        const float* __restrict__ W, const float* __restrict__ bias,
        float* __restrict__ C, int Ncols, int Mtiles) {
    int wv = threadIdx.x >> 5;
    int rowTile = blockIdx.x * 8 + wv;
    if (rowTile >= Mtiles) return;               // wave-uniform; EXEC stays all-1s for WMMA
    int lane = threadIdx.x & 31;
    int l16  = lane & 15;
    int half = lane >> 4;
    int row  = rowTile * 16 + l16;
    int col0 = blockIdx.y * 64 + l16;            // 4 consecutive 16-wide column tiles
    v8f acc0 = {0.f,0.f,0.f,0.f,0.f,0.f,0.f,0.f};
    v8f acc1 = acc0, acc2 = acc0, acc3 = acc0;
#pragma unroll
    for (int k0 = 0; k0 < KDIM; k0 += 32) {
        // A fragment: lanes 0-15 -> K = 0..7,16..23 ; lanes 16-31 -> K = 8..15,24..31
        const float* ar = A + (size_t)row * KDIM + k0 + half * 8;
        float4 a0 = *(const float4*)(ar);
        float4 a1 = *(const float4*)(ar + 4);
        float4 a2 = *(const float4*)(ar + 16);
        float4 a3 = *(const float4*)(ar + 20);
        v16bf a;
        a[0]=(__bf16)a0.x;  a[1]=(__bf16)a0.y;  a[2]=(__bf16)a0.z;  a[3]=(__bf16)a0.w;
        a[4]=(__bf16)a1.x;  a[5]=(__bf16)a1.y;  a[6]=(__bf16)a1.z;  a[7]=(__bf16)a1.w;
        a[8]=(__bf16)a2.x;  a[9]=(__bf16)a2.y;  a[10]=(__bf16)a2.z; a[11]=(__bf16)a2.w;
        a[12]=(__bf16)a3.x; a[13]=(__bf16)a3.y; a[14]=(__bf16)a3.z; a[15]=(__bf16)a3.w;
        // B fragments: lanes 0-15 -> K = 0..7,16..23 ; lanes 16-31 -> K = 8..15,24..31
        const float* wb = W + (size_t)(k0 + half * 8) * Ncols + col0;
        v16bf b0, b1, b2, b3;
#pragma unroll
        for (int kk = 0; kk < 16; kk++) {
            const float* wk = wb + (size_t)(kk < 8 ? kk : kk + 8) * Ncols;
            b0[kk] = (__bf16)wk[0];
            b1[kk] = (__bf16)wk[16];
            b2[kk] = (__bf16)wk[32];
            b3[kk] = (__bf16)wk[48];
        }
        acc0 = __builtin_amdgcn_wmma_f32_16x16x32_bf16(false, a, false, b0, (short)0, acc0, false, false);
        acc1 = __builtin_amdgcn_wmma_f32_16x16x32_bf16(false, a, false, b1, (short)0, acc1, false, false);
        acc2 = __builtin_amdgcn_wmma_f32_16x16x32_bf16(false, a, false, b2, (short)0, acc2, false, false);
        acc3 = __builtin_amdgcn_wmma_f32_16x16x32_bf16(false, a, false, b3, (short)0, acc3, false, false);
    }
    float bv0 = bias ? bias[col0]      : 0.f;
    float bv1 = bias ? bias[col0 + 16] : 0.f;
    float bv2 = bias ? bias[col0 + 32] : 0.f;
    float bv3 = bias ? bias[col0 + 48] : 0.f;
    // C layout: VGPR r -> row rowTile*16 + half*8 + r, col = l16 (+16/32/48 per tile)
    float* cp = C + (size_t)(rowTile * 16 + half * 8) * Ncols + col0;
#pragma unroll
    for (int r = 0; r < 8; r++) {
        float* cr = cp + (size_t)r * Ncols;
        cr[0]  = acc0[r] + bv0;
        cr[16] = acc1[r] + bv1;
        cr[32] = acc2[r] + bv2;
        cr[48] = acc3[r] + bv3;
    }
}

// ---------------- per-node attention scores a_src / a_dst ----------------
__global__ void k_scores(const float* __restrict__ hW, const float* __restrict__ atts,
                         const float* __restrict__ attd, float* __restrict__ as_,
                         float* __restrict__ ad_, int Cc) {
    int idx = blockIdx.x * blockDim.x + threadIdx.x;
    if (idx >= NN * 2) return;
    int n = idx >> 1, h = idx & 1;
    const float* rowp = hW + (size_t)n * 2 * Cc + h * Cc;
    float s1 = 0.f, s2 = 0.f;
    for (int c = 0; c < Cc; c++) {
        float v = rowp[c];
        s1 += v * atts[h * Cc + c];
        s2 += v * attd[h * Cc + c];
    }
    as_[idx] = s1; ad_[idx] = s2;
}

// ---------------- edge pass 1: alpha + segment max ----------------
__global__ void k_edge_alpha(const int* __restrict__ ei, const float* __restrict__ as_,
                             const float* __restrict__ ad_, float* __restrict__ alpha,
                             unsigned* __restrict__ amaxU) {
    int idx = blockIdx.x * blockDim.x + threadIdx.x;
    if (idx >= ETOT * 2) return;
    int e = idx >> 1, h = idx & 1;
    int s, d;
    if (e < EE) { s = ei[e]; d = ei[EE + e]; } else { s = d = e - EE; }
    float a = as_[s * 2 + h] + ad_[d * 2 + h];
    a = a > 0.f ? a : 0.2f * a;                 // leaky_relu 0.2
    alpha[idx] = a;
    atomicMax(&amaxU[d * 2 + h], f2ord(a));
}

// ---------------- edge pass 2: exp + segment sum ----------------
__global__ void k_edge_exp(const int* __restrict__ ei, float* __restrict__ alpha,
                           const unsigned* __restrict__ amaxU, float* __restrict__ denom) {
    int idx = blockIdx.x * blockDim.x + threadIdx.x;
    if (idx >= ETOT * 2) return;
    int e = idx >> 1, h = idx & 1;
    int d = (e < EE) ? ei[EE + e] : (e - EE);
    float ea = __expf(alpha[idx] - ord2f(amaxU[d * 2 + h]));
    alpha[idx] = ea;                             // reuse buffer as ea
    atomicAdd(&denom[d * 2 + h], ea);
}

// ---------------- edge pass 3: weighted message scatter ----------------
__global__ void k_edge_msg(const int* __restrict__ ei, const float* __restrict__ hW,
                           const float* __restrict__ ea, const float* __restrict__ denom,
                           float* __restrict__ gout, int hcs, int cs) {
    int idx = blockIdx.x * blockDim.x + threadIdx.x;   // grid sized exactly
    int e = idx >> hcs;
    int j = idx & ((1 << hcs) - 1);
    int h = j >> cs;
    int s, d;
    if (e < EE) { s = ei[e]; d = ei[EE + e]; } else { s = d = e - EE; }
    float att = ea[e * 2 + h] / (denom[d * 2 + h] + 1e-16f);
    atomicAdd(&gout[((size_t)d << hcs) + j], hW[((size_t)s << hcs) + j] * att);
}

// ---------------- finalize: concat/mean + bias + residual + relu ----------------
__global__ void k_finalize(const float* __restrict__ gout, const float* __restrict__ res,
                           const float* __restrict__ bias, float* __restrict__ hnew,
                           int mean_heads) {
    int idx = blockIdx.x * blockDim.x + threadIdx.x;   // N*128 exact
    int n = idx >> 7, j = idx & 127;
    float v;
    if (mean_heads) v = 0.5f * (gout[(size_t)n * 256 + j] + gout[(size_t)n * 256 + 128 + j]);
    else            v = gout[(size_t)n * 128 + j];
    v += bias[j] + res[idx];
    hnew[idx] = v > 0.f ? v : 0.f;
}

// ---------------- cluster assign: [N,64]@[64,8] + softmax over 8 ----------------
__global__ void k_ca2(const float* __restrict__ hid, const float* __restrict__ W2,
                      const float* __restrict__ b2, float* __restrict__ assign) {
    int idx = blockIdx.x * blockDim.x + threadIdx.x;   // N*8 exact (800000/256)
    int n = idx >> 3, cl = idx & 7;
    float acc = b2[cl];
    const float* hp = hid + (size_t)n * 64;
    for (int k = 0; k < 64; k++) acc += hp[k] * W2[k * 8 + cl];
    float mx = acc;
    for (int off = 4; off; off >>= 1) mx = fmaxf(mx, __shfl_xor(mx, off, 8));
    float ev = __expf(acc - mx);
    float s = ev;
    for (int off = 4; off; off >>= 1) s += __shfl_xor(s, off, 8);
    assign[idx] = ev / s;
}

// ---------------- pooled outer-product segment sum ----------------
__global__ __launch_bounds__(1024) void k_pool(const float* __restrict__ assign,
        const float* __restrict__ h, const int* __restrict__ batch,
        float* __restrict__ clusters) {
    int c = threadIdx.x >> 7;      // 0..7
    int dch = threadIdx.x & 127;   // 0..127
    int n0 = blockIdx.x * 64;
    float acc = 0.f;
    int curb = -1;
    for (int i = 0; i < 64; i++) {
        int n = n0 + i;
        if (n >= NN) break;
        int b = batch[n];
        if (b != curb) {
            if (curb >= 0) atomicAdd(&clusters[(size_t)curb * 1024 + c * 128 + dch], acc);
            curb = b; acc = 0.f;
        }
        acc += assign[(size_t)n * 8 + c] * h[(size_t)n * 128 + dch];
    }
    if (curb >= 0) atomicAdd(&clusters[(size_t)curb * 1024 + c * 128 + dch], acc);
}

// ---------------- max/min over clusters -> pooled [B,256] ----------------
__global__ void k_minmax(const float* __restrict__ clusters, float* __restrict__ pooled) {
    int idx = blockIdx.x * blockDim.x + threadIdx.x;   // 64*128 exact
    int b = idx >> 7, d = idx & 127;
    float mx = -3.4e38f, mn = 3.4e38f;
    for (int c = 0; c < 8; c++) {
        float v = clusters[(size_t)b * 1024 + c * 128 + d];
        mx = fmaxf(mx, v); mn = fminf(mn, v);
    }
    pooled[(size_t)b * 256 + d] = mx;
    pooled[(size_t)b * 256 + 128 + d] = mn;
}

// ---------------- head: dr linear + LN(16) + leaky(0.1) + classifier ----------------
__global__ __launch_bounds__(256) void k_head(const float* __restrict__ pooled,
        const float* __restrict__ drW, const float* __restrict__ drb,
        const float* __restrict__ lng, const float* __restrict__ lnb,
        const float* __restrict__ clsW, const float* __restrict__ clsb,
        float* __restrict__ out) {
    int t = blockIdx.x * blockDim.x + threadIdx.x;     // 64*16 = 1024
    int b = t >> 4, pd = t & 15;
    float acc = drb[pd];
    const float* pp = pooled + (size_t)b * 256;
    for (int k = 0; k < 256; k++) acc += pp[k] * drW[k * 16 + pd];
    float m = acc;
    for (int off = 8; off; off >>= 1) m += __shfl_xor(m, off, 16);
    m *= (1.f / 16.f);
    float d = acc - m;
    float v = d * d;
    for (int off = 8; off; off >>= 1) v += __shfl_xor(v, off, 16);
    v *= (1.f / 16.f);
    float g = d * rsqrtf(v + 1e-5f) * lng[pd] + lnb[pd];
    g = g > 0.f ? g : 0.1f * g;
    __shared__ float gs[256];
    gs[threadIdx.x] = g;
    __syncthreads();
    if (pd < 2) {
        float o = clsb[pd];
        int base = threadIdx.x & ~15;
        for (int k = 0; k < 16; k++) o += gs[base + k] * clsW[k * 2 + pd];
        out[b * 2 + pd] = o;
    }
}

extern "C" void kernel_launch(void* const* d_in, const int* in_sizes, int n_in,
                              void* d_out, int out_size, void* d_ws, size_t ws_size,
                              hipStream_t stream) {
    const float* x     = (const float*)d_in[0];
    const int*   ei    = (const int*)d_in[1];
    const int*   batch = (const int*)d_in[2];
    const float* W_in  = (const float*)d_in[3];
    const float* b_in  = (const float*)d_in[4];
    const float* ln_g  = (const float*)d_in[5];
    const float* ln_b  = (const float*)d_in[6];
    const float* g1_W  = (const float*)d_in[7];
    const float* g1_as = (const float*)d_in[8];
    const float* g1_ad = (const float*)d_in[9];
    const float* g1_b  = (const float*)d_in[10];
    const float* g2_W  = (const float*)d_in[11];
    const float* g2_as = (const float*)d_in[12];
    const float* g2_ad = (const float*)d_in[13];
    const float* g2_b  = (const float*)d_in[14];
    const float* r1_W  = (const float*)d_in[15];
    const float* r1_b  = (const float*)d_in[16];
    const float* r2_W  = (const float*)d_in[17];
    const float* r2_b  = (const float*)d_in[18];
    const float* ca_W1 = (const float*)d_in[19];
    const float* ca_b1 = (const float*)d_in[20];
    const float* ca_W2 = (const float*)d_in[21];
    const float* ca_b2 = (const float*)d_in[22];
    const float* dr_W  = (const float*)d_in[23];
    const float* dr_b  = (const float*)d_in[24];
    const float* drlng = (const float*)d_in[25];
    const float* drlnb = (const float*)d_in[26];
    const float* clsW  = (const float*)d_in[27];
    const float* clsb  = (const float*)d_in[28];
    float* out = (float*)d_out;

    float* ws = (float*)d_ws;
    size_t o = 0;
    float*    hA     = ws + o; o += (size_t)NN * 128;   // activations (in-place per layer)
    float*    hWb    = ws + o; o += (size_t)NN * 256;   // GEMM h@W output (reused as ca hidden)
    float*    res    = ws + o; o += (size_t)NN * 128;   // residual GEMM output
    float*    gout   = ws + o; o += (size_t)NN * 256;   // GAT aggregation output
    float*    ascr   = ws + o; o += (size_t)NN * 2;
    float*    adst   = ws + o; o += (size_t)NN * 2;
    unsigned* amaxU  = (unsigned*)(ws + o); o += (size_t)NN * 2;
    float*    denom  = ws + o; o += (size_t)NN * 2;
    float*    alpha  = ws + o; o += (size_t)ETOT * 2;   // alpha then ea
    float*    assign = ws + o; o += (size_t)NN * 8;
    float*    clus   = ws + o; o += (size_t)BB * 8 * 128;
    float*    pooled = ws + o; o += (size_t)BB * 256;
    (void)ws_size; (void)in_sizes; (void)n_in; (void)out_size;

    const int GX = (MT + 7) / 8;   // 782 blocks of 8 waves (wave = one 16-row tile)

    // input projection + LN + relu
    k_input<<<NN, 128, 0, stream>>>(x, W_in, b_in, ln_g, ln_b, hA);

    // ---- GAT layer 1 (concat, C=64, HC=128) ----
    k_gemm<128><<<dim3(GX, 2), 256, 0, stream>>>(hA, g1_W, nullptr, hWb, 128, MT);
    k_gemm<128><<<dim3(GX, 2), 256, 0, stream>>>(hA, r1_W, r1_b,   res, 128, MT);
    k_scores<<<(NN * 2 + 255) / 256, 256, 0, stream>>>(hWb, g1_as, g1_ad, ascr, adst, 64);
    k_fill<<<(NN * 2 + 255) / 256, 256, 0, stream>>>((float*)amaxU, 0.0f, NN * 2);
    k_fill<<<(NN * 2 + 255) / 256, 256, 0, stream>>>(denom, 0.0f, NN * 2);
    k_fill<<<(NN * 128) / 256, 256, 0, stream>>>(gout, 0.0f, NN * 128);
    k_edge_alpha<<<(ETOT * 2 + 255) / 256, 256, 0, stream>>>(ei, ascr, adst, alpha, amaxU);
    k_edge_exp<<<(ETOT * 2 + 255) / 256, 256, 0, stream>>>(ei, alpha, amaxU, denom);
    k_edge_msg<<<(ETOT * 128) / 256, 256, 0, stream>>>(ei, hWb, alpha, denom, gout, 7, 6);
    k_finalize<<<(NN * 128) / 256, 256, 0, stream>>>(gout, res, g1_b, hA, 0);

    // ---- GAT layer 2 (mean, C=128, HC=256) ----
    k_gemm<128><<<dim3(GX, 4), 256, 0, stream>>>(hA, g2_W, nullptr, hWb, 256, MT);
    k_gemm<128><<<dim3(GX, 2), 256, 0, stream>>>(hA, r2_W, r2_b,   res, 128, MT);
    k_scores<<<(NN * 2 + 255) / 256, 256, 0, stream>>>(hWb, g2_as, g2_ad, ascr, adst, 128);
    k_fill<<<(NN * 2 + 255) / 256, 256, 0, stream>>>((float*)amaxU, 0.0f, NN * 2);
    k_fill<<<(NN * 2 + 255) / 256, 256, 0, stream>>>(denom, 0.0f, NN * 2);
    k_fill<<<(NN * 256) / 256, 256, 0, stream>>>(gout, 0.0f, NN * 256);
    k_edge_alpha<<<(ETOT * 2 + 255) / 256, 256, 0, stream>>>(ei, ascr, adst, alpha, amaxU);
    k_edge_exp<<<(ETOT * 2 + 255) / 256, 256, 0, stream>>>(ei, alpha, amaxU, denom);
    k_edge_msg<<<(ETOT * 256) / 256, 256, 0, stream>>>(ei, hWb, alpha, denom, gout, 8, 7);
    k_finalize<<<(NN * 128) / 256, 256, 0, stream>>>(gout, res, g2_b, hA, 1);

    // ---- cluster assignment + pooling ----
    k_gemm<128><<<dim3(GX, 1), 256, 0, stream>>>(hA, ca_W1, ca_b1, hWb, 64, MT);  // [N,64]
    k_ca2<<<(NN * 8) / 256, 256, 0, stream>>>(hWb, ca_W2, ca_b2, assign);
    k_fill<<<(BB * 8 * 128) / 256, 256, 0, stream>>>(clus, 0.0f, BB * 8 * 128);
    k_pool<<<(NN + 63) / 64, 1024, 0, stream>>>(assign, hA, batch, clus);
    k_minmax<<<(BB * 128) / 256, 256, 0, stream>>>(clus, pooled);

    // ---- head ----
    k_head<<<(BB * 16) / 256, 256, 0, stream>>>(pooled, dr_W, dr_b, drlng, drlnb,
                                                clsW, clsb, out);
}
